// LocalPoolPointnetPPFusion_47261820125618
// MI455X (gfx1250) — matrix operand order repaired and checked by reading, not compiled, not run on hardware
//
#include <hip/hip_runtime.h>

typedef __attribute__((ext_vector_type(16))) __bf16 v16bf;
typedef __attribute__((ext_vector_type(8)))  __bf16 v8bf;
typedef __attribute__((ext_vector_type(8)))  float  v8f;

#define NBATCH 8
#define TPTS   8192
#define NPTS   (NBATCH * TPTS)   // 65536 points
#define HDIM   128
#define H2     256
#define RRES   128
#define NBINS  (RRES * RRES)     // 16384
#define EPSV   1e-3f

// ---------- helpers ----------
__device__ __forceinline__ unsigned short f32_to_bf16(float f) {
  unsigned u = __float_as_uint(f);
  u = (u + 0x7FFFu + ((u >> 16) & 1u)) >> 16;  // RNE
  return (unsigned short)u;
}
// monotone float<->uint key for atomicMax-based float max
__device__ __forceinline__ unsigned enc_key(float f) {
  unsigned u = __float_as_uint(f);
  return (u & 0x80000000u) ? ~u : (u | 0x80000000u);
}
__device__ __forceinline__ float dec_key(unsigned k) {
  unsigned u = (k & 0x80000000u) ? (k & 0x7FFFFFFFu) : ~k;
  return __uint_as_float(u);
}

// ---------- tiny utility kernels ----------
__global__ void k_cvt_bf16(const float* __restrict__ src, unsigned short* __restrict__ dst, int n) {
  int i = blockIdx.x * blockDim.x + threadIdx.x;
  if (i < n) dst[i] = f32_to_bf16(src[i]);
}
__global__ void k_fill_f32(float* __restrict__ p, float v, int n) {
  int i = blockIdx.x * blockDim.x + threadIdx.x;
  if (i < n) p[i] = v;
}
__global__ void k_fill_u32(unsigned* __restrict__ p, unsigned v, int n) {
  int i = blockIdx.x * blockDim.x + threadIdx.x;
  if (i < n) p[i] = v;
}

// ---------- plane index computation ----------
__global__ void k_idx(const float* __restrict__ p, int* __restrict__ idx) {
  int n = blockIdx.x * blockDim.x + threadIdx.x;
  if (n >= NPTS) return;
  const float inv = 1.0f / (1.0f + EPSV);
  float v3[3] = { p[n*3+0], p[n*3+1], p[n*3+2] };
  const int c0[3] = {0, 0, 1};  // PLANE_COLS = (0,2),(0,1),(1,2)
  const int c1[3] = {2, 1, 2};
#pragma unroll
  for (int pl = 0; pl < 3; pl++) {
    float u = v3[c0[pl]] * inv + 0.5f; u = fminf(fmaxf(u, 0.0f), 1.0f - EPSV);
    float w = v3[c1[pl]] * inv + 0.5f; w = fminf(fmaxf(w, 0.0f), 1.0f - EPSV);
    int gu = (int)floorf(u * (float)RRES);
    int gv = (int)floorf(w * (float)RRES);
    idx[pl * NPTS + n] = gu + RRES * gv;
  }
}

// ---------- input projection: netfull = p@wp + bp (+ p2@wp2 + bp2) ----------
__global__ void k_encode(const float* __restrict__ p, const float* __restrict__ p2,
                         const float* __restrict__ wp, const float* __restrict__ bp,
                         const float* __restrict__ wp2, const float* __restrict__ bp2,
                         int corr, float* __restrict__ netfull) {
  int t = blockIdx.x * blockDim.x + threadIdx.x;
  if (t >= NPTS * H2) return;
  int j = t & (H2 - 1);
  int n = t >> 8;
  float acc = bp[j];
#pragma unroll
  for (int d = 0; d < 3; d++) acc += p[n*3+d] * wp[d*H2 + j];
  if (corr) {
    acc += bp2[j];
#pragma unroll
    for (int d = 0; d < 3; d++) acc += p2[n*3+d] * wp2[d*H2 + j];
  }
  netfull[(size_t)n * H2 + j] = acc;
}

// ---------- pooling: scatter-max / gather ----------
__global__ void k_scatter_max(const float* __restrict__ net, const int* __restrict__ idx,
                              unsigned* __restrict__ bins) {
  int t = blockIdx.x * blockDim.x + threadIdx.x;
  if (t >= NPTS * HDIM) return;
  int c = t & (HDIM - 1);
  int n = t >> 7;
  int b = n / TPTS;
  int g = idx[n];
  atomicMax(&bins[((size_t)b * NBINS + g) * HDIM + c], enc_key(net[(size_t)n * HDIM + c]));
}
__global__ void k_gather_add(const unsigned* __restrict__ bins, const int* __restrict__ idx,
                             float* __restrict__ pooled) {
  int t = blockIdx.x * blockDim.x + threadIdx.x;
  if (t >= NPTS * HDIM) return;
  int c = t & (HDIM - 1);
  int n = t >> 7;
  int b = n / TPTS;
  int g = idx[n];
  pooled[(size_t)n * HDIM + c] += dec_key(bins[((size_t)b * NBINS + g) * HDIM + c]);
}

// ---------- scatter-mean rasterization ----------
__global__ void k_scatter_mean(const float* __restrict__ cf, const int* __restrict__ idx,
                               float* __restrict__ sums, float* __restrict__ cnt) {
  int t = blockIdx.x * blockDim.x + threadIdx.x;
  if (t >= NPTS * HDIM) return;
  int c = t & (HDIM - 1);
  int n = t >> 7;
  int b = n / TPTS;
  int g = idx[n];
  atomicAdd(&sums[((size_t)b * NBINS + g) * HDIM + c], cf[(size_t)n * HDIM + c]);
  if (c == 0) atomicAdd(&cnt[(size_t)b * NBINS + g], 1.0f);
}
// out[b][c][g] = sums[b][g][c] / max(cnt,1)
__global__ void k_plane_out(const float* __restrict__ sums, const float* __restrict__ cnt,
                            float* __restrict__ outp) {
  int t = blockIdx.x * blockDim.x + threadIdx.x;
  if (t >= NBATCH * HDIM * NBINS) return;
  int g = t & (NBINS - 1);
  int bc = t >> 14;
  int c = bc & (HDIM - 1);
  int b = bc >> 7;
  float cc = cnt[(size_t)b * NBINS + g];
  outp[t] = sums[((size_t)b * NBINS + g) * HDIM + c] / fmaxf(cc, 1.0f);
}

// ---------- fused ResnetBlockFC via WMMA ----------
// out[N,128] = x@ws + ( relu( relu(x)@w0 + b0 ) @ w1 + b1 ), x = [xA(:,0:128) | xB(:,0:128)]
__global__ __launch_bounds__(256) void k_resblock(
    const float* __restrict__ xA, int sA,
    const float* __restrict__ xB, int sB,
    const unsigned short* __restrict__ w0,   // [256][128] bf16
    const unsigned short* __restrict__ wsb,  // [256][128] bf16
    const unsigned short* __restrict__ w1,   // [128][128] bf16
    const float* __restrict__ b0,
    const float* __restrict__ b1,
    float* __restrict__ out) {
  __shared__ __align__(16) unsigned short lds_h[8][16][136];  // padded bf16 h-tile per wave
  const int wave  = threadIdx.x >> 5;
  const int lane  = threadIdx.x & 31;
  const int r0    = (blockIdx.x * 8 + wave) * 16;
  const int mrow  = lane & 15;
  const int halfk = lane >> 4;
  const int rbase = halfk * 8;

  v8f hacc[8], sacc[8];
#pragma unroll
  for (int i = 0; i < 8; i++) {
#pragma unroll
    for (int j = 0; j < 8; j++) { hacc[i][j] = 0.0f; sacc[i][j] = 0.0f; }
  }

  // GEMM1 over K=256: hacc += relu(x)@w0 ; sacc += x@ws
  for (int k0 = 0; k0 < H2; k0 += 32) {
    const float* src = (k0 < HDIM) ? xA : xB;
    const int stride = (k0 < HDIM) ? sA : sB;
    const int kk     = (k0 < HDIM) ? k0 : (k0 - HDIM);
    const float* rp = src + (size_t)(r0 + mrow) * stride + kk + halfk * 8;
    v16bf araw, arel;
#pragma unroll
    for (int i = 0; i < 8; i++) {
      float f0 = rp[i], f1 = rp[16 + i];
      araw[i]     = (__bf16)f0;
      araw[8 + i] = (__bf16)f1;
      arel[i]     = (__bf16)fmaxf(f0, 0.0f);
      arel[8 + i] = (__bf16)fmaxf(f1, 0.0f);
    }
    const unsigned short* w0r = w0  + (size_t)(k0 + halfk * 8) * HDIM + mrow;
    const unsigned short* wsr = wsb + (size_t)(k0 + halfk * 8) * HDIM + mrow;
#pragma unroll
    for (int nt = 0; nt < 8; nt++) {
      v16bf bm0, bms;
#pragma unroll
      for (int i = 0; i < 8; i++) {
        bm0[i]     = __builtin_bit_cast(__bf16, w0r[i * HDIM + nt * 16]);
        bm0[8 + i] = __builtin_bit_cast(__bf16, w0r[(16 + i) * HDIM + nt * 16]);
        bms[i]     = __builtin_bit_cast(__bf16, wsr[i * HDIM + nt * 16]);
        bms[8 + i] = __builtin_bit_cast(__bf16, wsr[(16 + i) * HDIM + nt * 16]);
      }
      hacc[nt] = __builtin_amdgcn_wmma_f32_16x16x32_bf16(false, arel, false, bm0, (short)0, hacc[nt], false, false);
      sacc[nt] = __builtin_amdgcn_wmma_f32_16x16x32_bf16(false, araw, false, bms, (short)0, sacc[nt], false, false);
    }
  }

  // h = relu(hacc + b0) -> LDS tile in A-feedable layout (same wave: LDS in-order, no barrier)
#pragma unroll
  for (int nt = 0; nt < 8; nt++) {
    int col = nt * 16 + mrow;
    float bias = b0[col];
#pragma unroll
    for (int r = 0; r < 8; r++) {
      float v = hacc[nt][r] + bias;
      lds_h[wave][rbase + r][col] = f32_to_bf16(fmaxf(v, 0.0f));
    }
  }

  // GEMM2 over K=128: sacc += h@w1
  for (int k0 = 0; k0 < HDIM; k0 += 32) {
    const __bf16* hp = reinterpret_cast<const __bf16*>(&lds_h[wave][mrow][k0 + halfk * 8]);
    v8bf lo = *reinterpret_cast<const v8bf*>(hp);
    v8bf hi = *reinterpret_cast<const v8bf*>(hp + 16);
    v16bf a;
#pragma unroll
    for (int i = 0; i < 8; i++) { a[i] = lo[i]; a[8 + i] = hi[i]; }
    const unsigned short* w1r = w1 + (size_t)(k0 + halfk * 8) * HDIM + mrow;
#pragma unroll
    for (int nt = 0; nt < 8; nt++) {
      v16bf bm;
#pragma unroll
      for (int i = 0; i < 8; i++) {
        bm[i]     = __builtin_bit_cast(__bf16, w1r[i * HDIM + nt * 16]);
        bm[8 + i] = __builtin_bit_cast(__bf16, w1r[(16 + i) * HDIM + nt * 16]);
      }
      sacc[nt] = __builtin_amdgcn_wmma_f32_16x16x32_bf16(false, a, false, bm, (short)0, sacc[nt], false, false);
    }
  }

  // out = sacc + b1
#pragma unroll
  for (int nt = 0; nt < 8; nt++) {
    int col = nt * 16 + mrow;
    float bias = b1[col];
#pragma unroll
    for (int r = 0; r < 8; r++) {
      out[(size_t)(r0 + rbase + r) * HDIM + col] = sacc[nt][r] + bias;
    }
  }
}

// ---------- final FC via WMMA: out[N,128] = x@w + b ----------
__global__ __launch_bounds__(256) void k_fc(
    const float* __restrict__ x,
    const unsigned short* __restrict__ w,  // [128][128] bf16
    const float* __restrict__ bias,
    float* __restrict__ out) {
  const int wave  = threadIdx.x >> 5;
  const int lane  = threadIdx.x & 31;
  const int r0    = (blockIdx.x * 8 + wave) * 16;
  const int mrow  = lane & 15;
  const int halfk = lane >> 4;
  const int rbase = halfk * 8;

  v8f acc[8];
#pragma unroll
  for (int i = 0; i < 8; i++) {
#pragma unroll
    for (int j = 0; j < 8; j++) acc[i][j] = 0.0f;
  }
  for (int k0 = 0; k0 < HDIM; k0 += 32) {
    const float* rp = x + (size_t)(r0 + mrow) * HDIM + k0 + halfk * 8;
    v16bf a;
#pragma unroll
    for (int i = 0; i < 8; i++) {
      a[i]     = (__bf16)rp[i];
      a[8 + i] = (__bf16)rp[16 + i];
    }
    const unsigned short* wr = w + (size_t)(k0 + halfk * 8) * HDIM + mrow;
#pragma unroll
    for (int nt = 0; nt < 8; nt++) {
      v16bf bm;
#pragma unroll
      for (int i = 0; i < 8; i++) {
        bm[i]     = __builtin_bit_cast(__bf16, wr[i * HDIM + nt * 16]);
        bm[8 + i] = __builtin_bit_cast(__bf16, wr[(16 + i) * HDIM + nt * 16]);
      }
      acc[nt] = __builtin_amdgcn_wmma_f32_16x16x32_bf16(false, a, false, bm, (short)0, acc[nt], false, false);
    }
  }
#pragma unroll
  for (int nt = 0; nt < 8; nt++) {
    int col = nt * 16 + mrow;
    float bv = bias[col];
#pragma unroll
    for (int r = 0; r < 8; r++) {
      out[(size_t)(r0 + rbase + r) * HDIM + col] = acc[nt][r] + bv;
    }
  }
}

// ---------- host orchestration ----------
extern "C" void kernel_launch(void* const* d_in, const int* in_sizes, int n_in,
                              void* d_out, int out_size, void* d_ws, size_t ws_size,
                              hipStream_t stream) {
  (void)in_sizes; (void)n_in; (void)out_size; (void)ws_size;
  const float* p       = (const float*)d_in[0];
  const float* p2      = (const float*)d_in[1];
  const float* wp      = (const float*)d_in[2];
  const float* bp      = (const float*)d_in[3];
  const float* wp2     = (const float*)d_in[4];
  const float* bp2     = (const float*)d_in[5];
  const float* blk_w0  = (const float*)d_in[6];
  const float* blk_b0  = (const float*)d_in[7];
  const float* blk_w1  = (const float*)d_in[8];
  const float* blk_b1  = (const float*)d_in[9];
  const float* blk_ws  = (const float*)d_in[10];
  const float* blkc_w0 = (const float*)d_in[11];
  const float* blkc_b0 = (const float*)d_in[12];
  const float* blkc_w1 = (const float*)d_in[13];
  const float* blkc_b1 = (const float*)d_in[14];
  const float* blkc_ws = (const float*)d_in[15];
  const float* fc_c_w  = (const float*)d_in[16];
  const float* fc_c_b  = (const float*)d_in[17];
  const float* fc_cc_w = (const float*)d_in[18];
  const float* fc_cc_b = (const float*)d_in[19];
  float* out = (float*)d_out;

  char* wsp = (char*)d_ws;
  size_t off = 0;
  auto alloc = [&](size_t bytes) -> void* {
    void* r = wsp + off;
    off += (bytes + 255) & ~(size_t)255;
    return r;
  };
  int*      idxbuf  = (int*)alloc((size_t)3 * NPTS * sizeof(int));
  float*    netfull = (float*)alloc((size_t)NPTS * H2 * sizeof(float));     // also reused for c-features
  float*    netA    = (float*)alloc((size_t)NPTS * HDIM * sizeof(float));
  float*    netB    = (float*)alloc((size_t)NPTS * HDIM * sizeof(float));
  float*    pooled  = (float*)alloc((size_t)NPTS * HDIM * sizeof(float));
  unsigned* bins    = (unsigned*)alloc((size_t)NBATCH * NBINS * HDIM * 4);  // max-keys OR f32 sums
  float*    cnt     = (float*)alloc((size_t)NBATCH * NBINS * sizeof(float));
  unsigned short* bw0  = (unsigned short*)alloc((size_t)5 * H2 * HDIM * 2);
  unsigned short* bw1  = (unsigned short*)alloc((size_t)5 * HDIM * HDIM * 2);
  unsigned short* bwss = (unsigned short*)alloc((size_t)5 * H2 * HDIM * 2);
  unsigned short* bcw0 = (unsigned short*)alloc((size_t)5 * H2 * HDIM * 2);
  unsigned short* bcw1 = (unsigned short*)alloc((size_t)5 * HDIM * HDIM * 2);
  unsigned short* bcws = (unsigned short*)alloc((size_t)5 * H2 * HDIM * 2);
  unsigned short* bfc  = (unsigned short*)alloc((size_t)HDIM * HDIM * 2);
  unsigned short* bfcc = (unsigned short*)alloc((size_t)HDIM * HDIM * 2);

  auto cvt = [&](const float* s, unsigned short* d, int n) {
    k_cvt_bf16<<<(n + 255) / 256, 256, 0, stream>>>(s, d, n);
  };
  cvt(blk_w0,  bw0,  5 * H2 * HDIM);
  cvt(blk_w1,  bw1,  5 * HDIM * HDIM);
  cvt(blk_ws,  bwss, 5 * H2 * HDIM);
  cvt(blkc_w0, bcw0, 5 * H2 * HDIM);
  cvt(blkc_w1, bcw1, 5 * HDIM * HDIM);
  cvt(blkc_ws, bcws, 5 * H2 * HDIM);
  cvt(fc_c_w,  bfc,  HDIM * HDIM);
  cvt(fc_cc_w, bfcc, HDIM * HDIM);

  k_idx<<<NPTS / 256, 256, 0, stream>>>(p, idxbuf);

  auto run_stream = [&](const unsigned short* w0s, const float* b0s,
                        const unsigned short* w1s, const float* b1s,
                        const unsigned short* wss,
                        const unsigned short* fcw, const float* fcb,
                        int corr, float* outbase) {
    k_encode<<<(NPTS * H2) / 256, 256, 0, stream>>>(p, p2, wp, bp, wp2, bp2, corr, netfull);
    // block 0: x = netfull[:,0:256]
    k_resblock<<<NPTS / 128, 256, 0, stream>>>(
        netfull, H2, netfull + HDIM, H2,
        w0s, wss, w1s, b0s, b1s, netA);
    float* cur = netA;
    float* nxt = netB;
    for (int i = 1; i < 5; i++) {
      k_fill_f32<<<(NPTS * HDIM) / 256, 256, 0, stream>>>(pooled, 0.0f, NPTS * HDIM);
      for (int pl = 0; pl < 3; pl++) {
        k_fill_u32<<<(NBATCH * NBINS * HDIM) / 256, 256, 0, stream>>>(bins, 0u, NBATCH * NBINS * HDIM);
        k_scatter_max<<<(NPTS * HDIM) / 256, 256, 0, stream>>>(cur, idxbuf + (size_t)pl * NPTS, bins);
        k_gather_add<<<(NPTS * HDIM) / 256, 256, 0, stream>>>(bins, idxbuf + (size_t)pl * NPTS, pooled);
      }
      k_resblock<<<NPTS / 128, 256, 0, stream>>>(
          cur, HDIM, pooled, HDIM,
          w0s + (size_t)i * H2 * HDIM, wss + (size_t)i * H2 * HDIM,
          w1s + (size_t)i * HDIM * HDIM,
          b0s + (size_t)i * HDIM, b1s + (size_t)i * HDIM, nxt);
      float* t = cur; cur = nxt; nxt = t;
    }
    // c = cur @ fcw + fcb  (reuse netfull as c-feature buffer; block0 input is consumed)
    float* cfeat = netfull;
    k_fc<<<NPTS / 128, 256, 0, stream>>>(cur, fcw, fcb, cfeat);
    for (int pl = 0; pl < 3; pl++) {
      k_fill_f32<<<(NBATCH * NBINS * HDIM) / 256, 256, 0, stream>>>((float*)bins, 0.0f, NBATCH * NBINS * HDIM);
      k_fill_f32<<<(NBATCH * NBINS + 255) / 256, 256, 0, stream>>>(cnt, 0.0f, NBATCH * NBINS);
      k_scatter_mean<<<(NPTS * HDIM) / 256, 256, 0, stream>>>(cfeat, idxbuf + (size_t)pl * NPTS, (float*)bins, cnt);
      k_plane_out<<<(NBATCH * HDIM * NBINS) / 256, 256, 0, stream>>>(
          (float*)bins, cnt, outbase + (size_t)pl * NBATCH * HDIM * NBINS);
    }
  };

  // geometry stream -> planes 0..2 ; articulation/corr stream -> planes 3..5
  run_stream(bw0,  blk_b0,  bw1,  blk_b1,  bwss, bfc,  fc_c_b,  0, out);
  run_stream(bcw0, blkc_b0, bcw1, blkc_b1, bcws, bfcc, fc_cc_b, 1, out + (size_t)3 * NBATCH * HDIM * NBINS);
}
